// LatentFeature_18116172054847
// MI455X (gfx1250) — matrix operand order repaired and checked by reading, not compiled, not run on hardware
//
#include <hip/hip_runtime.h>
#include <hip/hip_bf16.h>

// Problem constants (match the reference)
#define B_   32
#define N_   1024
#define CIN_ 3
#define H_   128
#define H2_  256

typedef __attribute__((ext_vector_type(16))) _Float16 v16h;
typedef __attribute__((ext_vector_type(8)))  _Float16 v8h;
typedef __attribute__((ext_vector_type(8)))  float    v8f;

#define LDK 40   // padded K stride in halves (80 B: 16B-aligned, conflict-free)

// ---------------------------------------------------------------------------
// Batched GEMM on the CDNA5 WMMA path, LDS-staged.
//   C[b][m,n] = (sum_k A[b][m,k] * B[b][k,n] + bias[m]) * colscale[b][n]
// transA: A[m,k] read at A[k*lda + m] (used for energy = y^T y).
// Block = 256 threads = 8 waves; block tile 128(M) x 64(N); K-step 32.
// f32 -> f16 conversion happens once during LDS staging; fragments are pure
// ds_load_b128 in the ISA 7.12.2 layouts; f32 accumulation in WMMA.
// All M are multiples of 128 and K multiples of 32 for every call site.
// ---------------------------------------------------------------------------
__global__ __launch_bounds__(256) void k_gemm_wmma(
    const float* __restrict__ A, const float* __restrict__ Bm, float* __restrict__ C,
    int M, int K, int lda, int ldb, int ldc,
    size_t sA, size_t sB, size_t sC, int transA,
    const float* __restrict__ bias, const float* __restrict__ colscale)
{
    __shared__ _Float16 As[128 * LDK];   // As[m][k], K-contiguous
    __shared__ _Float16 Bs[64 * LDK];    // Bs[n][k], K-contiguous

    const int tid  = threadIdx.x;
    const int lane = tid & 31;
    const int wave = tid >> 5;
    const int hh   = lane >> 4;          // K-half selector
    const int r    = lane & 15;          // row/col within half
    const int m0b  = blockIdx.y * 128;   // block M origin
    const int n0b  = blockIdx.x * 64;    // block N origin
    const int b    = blockIdx.z;

    const float* Ab = A  + (size_t)b * sA;
    const float* Bb = Bm + (size_t)b * sB;
    float*       Cb = C  + (size_t)b * sC;

    union H4 { _Float16 h[4]; uint2 u2; };

    v8f acc[4] = {};
    for (int k0 = 0; k0 < K; k0 += 32) {
        __syncthreads();   // protect LDS tiles from previous iteration's readers

        // ---- stage A tile: 128 x 32 f32 -> f16 ----
        if (transA) {
            // memory view: 32 k-rows x 128 m-cols, contiguous in m
            #pragma unroll
            for (int i = 0; i < 4; ++i) {
                const int f   = tid + 256 * i;         // float4 slot, 0..1023
                const int kr  = f >> 5;                // 0..31
                const int mc4 = f & 31;                // *4 = m col
                const float* p = Ab + (size_t)(k0 + kr) * lda + m0b + 4 * mc4;
                const float4 v = *(const float4*)p;
                if (k0 + 32 < K) __builtin_prefetch(p + (size_t)32 * lda, 0, 1);
                As[(4 * mc4 + 0) * LDK + kr] = (_Float16)v.x;
                As[(4 * mc4 + 1) * LDK + kr] = (_Float16)v.y;
                As[(4 * mc4 + 2) * LDK + kr] = (_Float16)v.z;
                As[(4 * mc4 + 3) * LDK + kr] = (_Float16)v.w;
            }
        } else {
            // memory view: 128 m-rows x 32 k-cols, contiguous in k
            #pragma unroll
            for (int i = 0; i < 4; ++i) {
                const int f   = tid + 256 * i;         // float4 slot, 0..1023
                const int row = f >> 3;                // 0..127
                const int c4  = f & 7;                 // *4 = k col
                const float* p = Ab + (size_t)(m0b + row) * lda + k0 + 4 * c4;
                const float4 v = *(const float4*)p;
                if (k0 + 32 < K) __builtin_prefetch(p + 32, 0, 1);
                H4 t;
                t.h[0] = (_Float16)v.x; t.h[1] = (_Float16)v.y;
                t.h[2] = (_Float16)v.z; t.h[3] = (_Float16)v.w;
                *(uint2*)&As[row * LDK + 4 * c4] = t.u2;   // 8B-aligned
            }
        }

        // ---- stage B tile: 32 x 64 f32 -> f16, stored transposed Bs[n][k] ----
        #pragma unroll
        for (int i = 0; i < 2; ++i) {
            const int f   = tid + 256 * i;             // float4 slot, 0..511
            const int kr  = f >> 4;                    // 0..31
            const int nc4 = f & 15;                    // *4 = n col
            const float* p = Bb + (size_t)(k0 + kr) * ldb + n0b + 4 * nc4;
            const float4 v = *(const float4*)p;
            if (k0 + 32 < K) __builtin_prefetch(p + (size_t)32 * ldb, 0, 1);
            Bs[(4 * nc4 + 0) * LDK + kr] = (_Float16)v.x;
            Bs[(4 * nc4 + 1) * LDK + kr] = (_Float16)v.y;
            Bs[(4 * nc4 + 2) * LDK + kr] = (_Float16)v.z;
            Bs[(4 * nc4 + 3) * LDK + kr] = (_Float16)v.w;
        }

        __syncthreads();

        // ---- fragments from LDS (ISA 7.12.2 layouts), 4 WMMA per K-step ----
        // A 16x32: lane half hh needs k = 8*hh+0..7 (elems 0..7) and
        //          k = 16+8*hh+0..7 (elems 8..15): two b128 reads.
        const _Float16* arow = &As[(wave * 16 + r) * LDK + 8 * hh];
        const v8h alo = *(const v8h*)(arow);
        const v8h ahi = *(const v8h*)(arow + 16);
        const v16h a = __builtin_shufflevector(alo, ahi,
            0, 1, 2, 3, 4, 5, 6, 7, 8, 9, 10, 11, 12, 13, 14, 15);

        #pragma unroll
        for (int t = 0; t < 4; ++t) {
            // B 32x16: lane half hh needs k = 8*hh+0..7 and 16+8*hh+0..7 at fixed n.
            const _Float16* brow = &Bs[(t * 16 + r) * LDK + 8 * hh];
            const v8h blo = *(const v8h*)(brow);
            const v8h bhi = *(const v8h*)(brow + 16);
            const v16h bf = __builtin_shufflevector(blo, bhi,
                0, 1, 2, 3, 4, 5, 6, 7, 8, 9, 10, 11, 12, 13, 14, 15);
            acc[t] = __builtin_amdgcn_wmma_f32_16x16x32_f16(
                false, a, false, bf, (short)0, acc[t], false, false);
        }
    }

    // Epilogue: optional bias (per output row) and column scale (per batch/col)
    #pragma unroll
    for (int t = 0; t < 4; ++t) {
        const int nn = n0b + t * 16 + r;
        const float cs = colscale ? colscale[(size_t)b * N_ + nn] : 1.0f;
        #pragma unroll
        for (int i = 0; i < 8; ++i) {
            const int mm = m0b + wave * 16 + (hh << 3) + i;  // C VGPR i: M=8*hh+i
            float val = acc[t][i];
            if (bias) val += bias[mm];
            Cb[(size_t)mm * ldc + nn] = val * cs;
        }
    }
}

// ---------------------------------------------------------------------------
// K=3 input conv: out[b,o,n] = sum_c W_l1[o,c] * x[b,c,n]
// ---------------------------------------------------------------------------
__global__ __launch_bounds__(256) void k_conv_l1(
    const float* __restrict__ x, const float* __restrict__ W, float* __restrict__ out)
{
    const unsigned idx = blockIdx.x * 256u + threadIdx.x;  // B*H*N
    const int n = idx & (N_ - 1);
    const int o = (idx >> 10) & (H_ - 1);
    const int b = idx >> 17;
    const float* xb = x + (size_t)b * CIN_ * N_;
    out[idx] = W[o * 3 + 0] * xb[n] + W[o * 3 + 1] * xb[N_ + n] + W[o * 3 + 2] * xb[2 * N_ + n];
}

// ---------------------------------------------------------------------------
// In-place instance norm (over N=1024) + ReLU. One 256-thread block per row.
// ---------------------------------------------------------------------------
__global__ __launch_bounds__(256) void k_instnorm_relu(float* __restrict__ data)
{
    __shared__ float s1[256], s2[256];
    float* p = data + (size_t)blockIdx.x * N_;
    float v[4], sum = 0.f, sumsq = 0.f;
    #pragma unroll
    for (int i = 0; i < 4; ++i) {
        v[i] = p[threadIdx.x + 256 * i];
        sum += v[i]; sumsq += v[i] * v[i];
    }
    s1[threadIdx.x] = sum; s2[threadIdx.x] = sumsq; __syncthreads();
    for (int off = 128; off > 0; off >>= 1) {
        if (threadIdx.x < off) {
            s1[threadIdx.x] += s1[threadIdx.x + off];
            s2[threadIdx.x] += s2[threadIdx.x + off];
        }
        __syncthreads();
    }
    const float mean = s1[0] * (1.0f / N_);
    const float var  = s2[0] * (1.0f / N_) - mean * mean;
    const float inv  = rsqrtf(var + 1e-5f);
    #pragma unroll
    for (int i = 0; i < 4; ++i) {
        const float t = (v[i] - mean) * inv;
        p[threadIdx.x + 256 * i] = t > 0.f ? t : 0.f;
    }
}

// ---------------------------------------------------------------------------
// sq[b,n] = |a[b,n,:]|^2 where a[b,n,d] = x[b,d,n]
// ---------------------------------------------------------------------------
__global__ __launch_bounds__(256) void k_sqnorm(const float* __restrict__ x, float* __restrict__ sq)
{
    const unsigned idx = blockIdx.x * 256u + threadIdx.x;  // B*N
    const int n = idx & (N_ - 1), b = idx >> 10;
    const float* xb = x + (size_t)b * CIN_ * N_;
    const float a0 = xb[n], a1 = xb[N_ + n], a2 = xb[2 * N_ + n];
    sq[idx] = a0 * a0 + a1 * a1 + a2 * a2;
}

// ---------------------------------------------------------------------------
// dist[b,m,n] = safe_sqrt(sq[m] + sq[n] - 2 a.a)
// ---------------------------------------------------------------------------
__global__ __launch_bounds__(256) void k_dist(
    const float* __restrict__ x, const float* __restrict__ sq, float* __restrict__ dist)
{
    const unsigned idx = blockIdx.x * 256u + threadIdx.x;  // B*N*N
    const int n = idx & (N_ - 1);
    const int m = (idx >> 10) & (N_ - 1);
    const int b = idx >> 20;
    const float* xb = x + (size_t)b * CIN_ * N_;
    const float dot = xb[m] * xb[n] + xb[N_ + m] * xb[N_ + n] + xb[2 * N_ + m] * xb[2 * N_ + n];
    const float d2 = sq[b * N_ + m] + sq[b * N_ + n] - 2.0f * dot;
    dist[idx] = d2 > 0.f ? sqrtf(d2) : 0.f;
}

// ---------------------------------------------------------------------------
// Row softmax over the last dim (N=1024), in place. One block per (b, n) row.
// ---------------------------------------------------------------------------
__global__ __launch_bounds__(256) void k_softmax_row(float* __restrict__ att)
{
    __shared__ float sh[256];
    float* p = att + (size_t)blockIdx.x * N_;
    float v[4], mx = -3.4e38f;
    #pragma unroll
    for (int i = 0; i < 4; ++i) { v[i] = p[threadIdx.x + 256 * i]; mx = fmaxf(mx, v[i]); }
    sh[threadIdx.x] = mx; __syncthreads();
    for (int off = 128; off > 0; off >>= 1) {
        if (threadIdx.x < off) sh[threadIdx.x] = fmaxf(sh[threadIdx.x], sh[threadIdx.x + off]);
        __syncthreads();
    }
    mx = sh[0]; __syncthreads();
    float sum = 0.f;
    #pragma unroll
    for (int i = 0; i < 4; ++i) { v[i] = __expf(v[i] - mx); sum += v[i]; }
    sh[threadIdx.x] = sum; __syncthreads();
    for (int off = 128; off > 0; off >>= 1) {
        if (threadIdx.x < off) sh[threadIdx.x] += sh[threadIdx.x + off];
        __syncthreads();
    }
    const float inv = 1.0f / sh[0];
    #pragma unroll
    for (int i = 0; i < 4; ++i) p[threadIdx.x + 256 * i] = v[i] * inv;
}

// ---------------------------------------------------------------------------
// scale[b,m] = 1 / ((1e-9 + sum_n att[b,n,m]) * sqrt(H))  (column L1 renorm)
// ---------------------------------------------------------------------------
__global__ __launch_bounds__(256) void k_colscale(
    const float* __restrict__ att, float* __restrict__ scale)
{
    const unsigned idx = blockIdx.x * 256u + threadIdx.x;  // B*N
    const int m = idx & (N_ - 1), b = idx >> 10;
    const float* p = att + (size_t)b * N_ * N_ + m;
    float s = 0.f;
    for (int n = 0; n < N_; ++n) s += p[(size_t)n * N_];
    scale[idx] = 1.0f / ((1e-9f + s) * 11.313708498984761f /* sqrt(128) */);
}

// ---------------------------------------------------------------------------
// out[b, layer*H + c] = max_n feat[b,c,n]. One block per (b, c) row.
// ---------------------------------------------------------------------------
__global__ __launch_bounds__(256) void k_rowmax(
    const float* __restrict__ feat, float* __restrict__ out, int layer)
{
    __shared__ float sh[256];
    const int b = blockIdx.x >> 7, c = blockIdx.x & (H_ - 1);
    const float* p = feat + (size_t)blockIdx.x * N_;
    float mx = -3.4e38f;
    #pragma unroll
    for (int i = 0; i < 4; ++i) mx = fmaxf(mx, p[threadIdx.x + 256 * i]);
    sh[threadIdx.x] = mx; __syncthreads();
    for (int off = 128; off > 0; off >>= 1) {
        if (threadIdx.x < off) sh[threadIdx.x] = fmaxf(sh[threadIdx.x], sh[threadIdx.x + off]);
        __syncthreads();
    }
    if (threadIdx.x == 0) out[(size_t)b * (4 * H_) + layer * H_ + c] = sh[0];
}

// ---------------------------------------------------------------------------
static inline void gemm(hipStream_t s,
                        const float* A, const float* Bm, float* C,
                        int M, int K, int lda, int ldb, int ldc,
                        size_t sA, size_t sB, size_t sC, int transA,
                        const float* bias, const float* colscale)
{
    dim3 grid(N_ / 64, M / 128, B_);
    k_gemm_wmma<<<grid, 256, 0, s>>>(A, Bm, C, M, K, lda, ldb, ldc,
                                     sA, sB, sC, transA, bias, colscale);
}

extern "C" void kernel_launch(void* const* d_in, const int* in_sizes, int n_in,
                              void* d_out, int out_size, void* d_ws, size_t ws_size,
                              hipStream_t stream)
{
    const float* x    = (const float*)d_in[0];
    const float* W_l1 = (const float*)d_in[1];
    const float* W_l2 = (const float*)d_in[2];
    const float* W_d1 = (const float*)d_in[3];
    const float* W_d2 = (const float*)d_in[4];
    const float* Wqk  = (const float*)d_in[5];
    const float* Wv   = (const float*)d_in[6];
    const float* bv   = (const float*)d_in[7];
    float* out = (float*)d_out;

    // Workspace carve (all fp32). att (128 MB) stays L2-resident (192 MB L2).
    const size_t FEAT = (size_t)B_ * H_ * N_;
    float* ws    = (float*)d_ws;
    float* feat0 = ws;  ws += FEAT;
    float* feat1 = ws;  ws += FEAT;
    float* ybuf  = ws;  ws += FEAT;
    float* vbuf  = ws;  ws += FEAT;
    float* t256  = ws;  ws += (size_t)B_ * H2_ * N_;
    float* sq    = ws;  ws += (size_t)B_ * N_;
    float* scale = ws;  ws += (size_t)B_ * N_;
    float* att   = ws;

    // One offset self-attention layer (q/k tied => energy = y^T y, symmetric)
    auto sa = [&](const float* fin, float* fout, int layer) {
        const float* Wq  = Wqk + (size_t)layer * H_ * H_;
        const float* Wvl = Wv  + (size_t)layer * H_ * H_;
        const float* bvl = bv  + (size_t)layer * H_;
        // y = Wqk * x ; xv = Wv * x + bv
        gemm(stream, Wq,  fin, ybuf, H_, H_, H_, N_, N_, 0, FEAT, FEAT, 0, nullptr, nullptr);
        gemm(stream, Wvl, fin, vbuf, H_, H_, H_, N_, N_, 0, FEAT, FEAT, 0, bvl,     nullptr);
        // energy[n,m] = sum_c y[c,n] y[c,m]  (transA view of y)
        gemm(stream, ybuf, ybuf, att, N_, H_, N_, N_, N_,
             FEAT, FEAT, (size_t)N_ * N_, 1, nullptr, nullptr);
        k_softmax_row<<<B_ * N_, 256, 0, stream>>>(att);
        k_colscale<<<(B_ * N_) / 256, 256, 0, stream>>>(att, scale);
        // out[c,m] = (sum_n xv[c,n] att[n,m]) * scale[m]
        gemm(stream, vbuf, att, fout, H_, N_, N_, N_, N_,
             FEAT, (size_t)N_ * N_, FEAT, 0, nullptr, scale);
        k_rowmax<<<B_ * H_, 256, 0, stream>>>(fout, out, layer);
    };

    // ---- point-feature branch: h = relu(IN(W_l2 relu(IN(W_l1 x)))) ----
    k_conv_l1<<<FEAT / 256, 256, 0, stream>>>(x, W_l1, feat0);
    k_instnorm_relu<<<B_ * H_, 256, 0, stream>>>(feat0);
    gemm(stream, W_l2, feat0, feat1, H_, H_, H_, N_, N_, 0, FEAT, FEAT, 0, nullptr, nullptr);
    k_instnorm_relu<<<B_ * H_, 256, 0, stream>>>(feat1);

    sa(feat1, feat0, 0);   // x1
    sa(feat0, feat1, 1);   // x2

    // ---- distance-key branch (reuses att buffer for the distance matrix) ----
    k_sqnorm<<<(B_ * N_) / 256, 256, 0, stream>>>(x, sq);
    k_dist<<<(unsigned)((size_t)B_ * N_ * N_ / 256), 256, 0, stream>>>(x, sq, att);
    gemm(stream, W_d1, att, t256, H2_, N_, N_, N_, N_,
         0, (size_t)N_ * N_, (size_t)H2_ * N_, 0, nullptr, nullptr);
    k_instnorm_relu<<<B_ * H2_, 256, 0, stream>>>(t256);
    gemm(stream, W_d2, t256, feat0, H_, H2_, H2_, N_, N_,
         0, (size_t)H2_ * N_, FEAT, 0, nullptr, nullptr);

    sa(feat0, feat1, 2);   // x3
    sa(feat1, feat0, 3);   // x4
}
